// CrossAttention_63299228009015
// MI455X (gfx1250) — matrix operand order repaired and checked by reading, not compiled, not run on hardware
//
#include <hip/hip_runtime.h>
#include <hip/hip_bf16.h>

typedef __attribute__((ext_vector_type(16))) _Float16 v16h;
typedef __attribute__((ext_vector_type(8)))  _Float16 v8h;
typedef __attribute__((ext_vector_type(4)))  _Float16 v4h;
typedef __attribute__((ext_vector_type(8)))  float    v8f;
typedef __attribute__((ext_vector_type(4)))  float    v4f;

#define WMMA_F32_F16(a, b, c) \
    __builtin_amdgcn_wmma_f32_16x16x32_f16(false, (a), false, (b), (short)0, (c), false, false)

// ---------------------------------------------------------------------------
// CDNA5 async global->LDS path (ASYNCcnt-tracked, no VGPR round trip).
// Probe result: __builtin_amdgcn_global_load_async_to_lds_b128 takes
// (int4 addrspace(1)*, int4 addrspace(3)*, imm offset, imm cpol).
// Compile-safe fallback to a two-phase register copy if the builtin is absent.
// ---------------------------------------------------------------------------
#if __has_builtin(__builtin_amdgcn_global_load_async_to_lds_b128)
#define HAS_ASYNC_B128 1
typedef int v4i_vs __attribute__((__vector_size__(4 * sizeof(int))));
typedef __attribute__((address_space(1))) v4i_vs gv4i;
typedef __attribute__((address_space(3))) v4i_vs lv4i;
#else
#define HAS_ASYNC_B128 0
#endif

#if __has_builtin(__builtin_amdgcn_s_wait_asynccnt)
#define WAIT_ASYNC() __builtin_amdgcn_s_wait_asynccnt(0)
#else
#define WAIT_ASYNC() asm volatile("s_wait_asynccnt 0x0" ::: "memory")
#endif

// ---------------------------------------------------------------------------
// Fragment loaders (wave32, 16x16x32 f16 WMMA).
// A (16x32 f16): lane L holds row L&15; halves 0..7 = k in [8*(L>>4), +8),
//                halves 8..15 = k in [16 + 8*(L>>4), +8).
// B (32x16 f16): lane L holds column L&15 of the *transposed* operand
//                (row n of Bt where Bt[n][k] = B[k][n]);
//                halves 0..7 = k in [8*(L>>4), +8), halves 8..15 = k in [16 + 8*(L>>4), +8).
// `ld` is the row stride in halves; ld*2 must be a multiple of 16 bytes.
// ---------------------------------------------------------------------------
__device__ __forceinline__ v16h load_a16(const _Float16* base, int ld) {
    const int lane = threadIdx.x & 31;
    const _Float16* p = base + (lane & 15) * ld + ((lane >> 4) << 3);
    v8h lo = *(const v8h*)(p);
    v8h hi = *(const v8h*)(p + 16);
    v16h a;
#pragma unroll
    for (int i = 0; i < 8; ++i) { a[i] = lo[i]; a[i + 8] = hi[i]; }
    return a;
}

__device__ __forceinline__ v16h load_b16(const _Float16* base, int ld) {
    const int lane = threadIdx.x & 31;
    const _Float16* p = base + (lane & 15) * ld + ((lane >> 4) << 3);
    v8h lo = *(const v8h*)(p);
    v8h hi = *(const v8h*)(p + 16);
    v16h b;
#pragma unroll
    for (int i = 0; i < 8; ++i) { b[i] = lo[i]; b[i + 8] = hi[i]; }
    return b;
}

// ---------------------------------------------------------------------------
// Tile movers. 128 threads per block. All addresses 16B-aligned.
// ---------------------------------------------------------------------------
#define PLD 48  // LDS row stride (halves) for 32-wide k tiles; 96B, 16B-aligned
#define ALD 72  // LDS row stride (halves) for 64-wide tiles; 144B, 16B-aligned

// ROWS x 32 fp32 -> f16 LDS (stride PLD). CHUNKS = ROWS/16 float4's per thread.
template <int CHUNKS>
__device__ __forceinline__ void tile_f32_to_lds(const float* __restrict__ g,
                                                size_t ldg, _Float16* l, int t) {
    v4f tmp[CHUNKS];
#pragma unroll
    for (int i = 0; i < CHUNKS; ++i) {
        int lin = i * 128 + t;
        int r = lin >> 3, c = (lin & 7) << 2;
        tmp[i] = *(const v4f*)(g + (size_t)r * ldg + c);
    }
#pragma unroll
    for (int i = 0; i < CHUNKS; ++i) {
        int lin = i * 128 + t;
        int r = lin >> 3, c = (lin & 7) << 2;
        *(v4h*)(l + r * PLD + c) = __builtin_convertvector(tmp[i], v4h);
    }
}

// f16 tile -> LDS in 16B chunks. RSHIFT: chunks-per-row = 1<<RSHIFT.
// Covers (CHUNKS*128 >> RSHIFT) rows of (8 << RSHIFT) halves.
template <int CHUNKS, int RSHIFT>
__device__ __forceinline__ void tile_f16_issue(const _Float16* __restrict__ g,
                                               size_t ldg, _Float16* l, int ldl,
                                               int t) {
#if HAS_ASYNC_B128
#pragma unroll
    for (int i = 0; i < CHUNKS; ++i) {
        int lin = i * 128 + t;
        int r = lin >> RSHIFT, c = (lin & ((1 << RSHIFT) - 1)) << 3;
        __builtin_amdgcn_global_load_async_to_lds_b128(
            (gv4i*)(g + (size_t)r * ldg + c),
            (lv4i*)(l + r * ldl + c), 0, 0);
    }
#else
    v8h tmp[CHUNKS];
#pragma unroll
    for (int i = 0; i < CHUNKS; ++i) {
        int lin = i * 128 + t;
        int r = lin >> RSHIFT, c = (lin & ((1 << RSHIFT) - 1)) << 3;
        tmp[i] = *(const v8h*)(g + (size_t)r * ldg + c);
    }
#pragma unroll
    for (int i = 0; i < CHUNKS; ++i) {
        int lin = i * 128 + t;
        int r = lin >> RSHIFT, c = (lin & ((1 << RSHIFT) - 1)) << 3;
        *(v8h*)(l + r * ldl + c) = tmp[i];
    }
#endif
}

__device__ __forceinline__ void tile_f16_wait() {
#if HAS_ASYNC_B128
    WAIT_ASYNC();
#endif
}

// ---------------------------------------------------------------------------
// Projection GEMM:  y[r][e] = sum_k A[r][k] * W[e][k]   (torch Linear, no bias)
// A: fp32 [8192 x 1024] row-major, W: fp32 [1024 x 1024] row-major.
// Output f16, scaled, in per-head layout:
//   transposed==0:  out[((b*16+h)*2048 + n)*64 + d]       (Q, K)
//   transposed==1:  out[((b*16+h)*64   + d)*2048 + n]     (V^T)
// Block: 128 threads (4 waves). Tile: 128 rows x 64 cols, k-chunk 32.
// Wave w owns rows [32w, 32w+32): 2 A-frags x 4 B-frags = 8 WMMA per chunk.
// ---------------------------------------------------------------------------
__global__ void proj_kernel(const float* __restrict__ A,
                            const float* __restrict__ W,
                            _Float16* __restrict__ out,
                            float scale, int transposed) {
    __shared__ __align__(16) _Float16 aT[128 * PLD];
    __shared__ __align__(16) _Float16 wT[64 * PLD];

    const int t    = threadIdx.x;
    const int wave = t >> 5;
    const int lane = t & 31;
    const int row0 = blockIdx.x * 128;
    const int col0 = blockIdx.y * 64;

    v8f acc[2][4] = {};

    for (int k0 = 0; k0 < 1024; k0 += 32) {
        __syncthreads();
        tile_f32_to_lds<8>(A + (size_t)row0 * 1024 + k0, 1024, aT, t);
        tile_f32_to_lds<4>(W + (size_t)col0 * 1024 + k0, 1024, wT, t);
        __syncthreads();

        v16h a0 = load_a16(&aT[(wave * 32) * PLD], PLD);
        v16h a1 = load_a16(&aT[(wave * 32 + 16) * PLD], PLD);
#pragma unroll
        for (int tt = 0; tt < 4; ++tt) {
            v16h b = load_b16(&wT[(tt * 16) * PLD], PLD);
            acc[0][tt] = WMMA_F32_F16(a0, b, acc[0][tt]);
            acc[1][tt] = WMMA_F32_F16(a1, b, acc[1][tt]);
        }
    }

    // Epilogue: C layout -> per-head global layout.
    const int half = lane >> 4;
    const int nn   = lane & 15;
#pragma unroll
    for (int sub = 0; sub < 2; ++sub) {
#pragma unroll
        for (int tt = 0; tt < 4; ++tt) {
#pragma unroll
            for (int v = 0; v < 8; ++v) {
                int r = row0 + wave * 32 + sub * 16 + v + 8 * half;  // [0, 8192)
                int e = col0 + tt * 16 + nn;                         // [0, 1024)
                int b = r >> 11, n = r & 2047;
                int h = e >> 6,  d = e & 63;
                size_t off = transposed
                    ? (((size_t)(b * 16 + h) * 64 + d) * 2048 + n)
                    : (((size_t)(b * 16 + h) * 2048 + n) * 64 + d);
                out[off] = (_Float16)(acc[sub][tt][v] * scale);
            }
        }
    }
}

// ---------------------------------------------------------------------------
// Flash attention. Grid: (N/64, B*H), block 128 (4 waves).
// Wave w owns query rows [q0 + 16w, +16). Streams 64-key tiles of K and V^T
// through LDS; online softmax in f32; P re-laid out via LDS for the PV WMMA.
// q is pre-scaled by 1/sqrt(64) at projection time.
// ---------------------------------------------------------------------------
__global__ void attn_kernel(const _Float16* __restrict__ q,
                            const _Float16* __restrict__ k,
                            const _Float16* __restrict__ vt,
                            _Float16* __restrict__ mid) {
    __shared__ __align__(16) _Float16 kT[64 * ALD];        // K tile  [key][d]
    __shared__ __align__(16) _Float16 vtT[64 * ALD];       // V^T tile [d][key]
    __shared__ __align__(16) _Float16 pT[4][16 * ALD];     // per-wave P tile

    const int t    = threadIdx.x;
    const int wave = t >> 5;
    const int lane = t & 31;
    const int bh   = blockIdx.y;        // b*16 + h
    const int q0   = blockIdx.x * 64;

    // Q fragments for this wave, loaded once from global (rows are 128B).
    const _Float16* qbase = q + ((size_t)bh * 2048 + q0 + wave * 16) * 64;
    const v16h aq0 = load_a16(qbase, 64);        // k-slice 0..31
    const v16h aq1 = load_a16(qbase + 32, 64);   // k-slice 32..63

    const _Float16* kbase  = k  + (size_t)bh * 2048 * 64;
    const _Float16* vtbase = vt + (size_t)bh * 64 * 2048;

    float mrow[8], lrow[8];
    v8f o[4] = {};
#pragma unroll
    for (int v = 0; v < 8; ++v) { mrow[v] = -1e30f; lrow[v] = 0.f; }

    for (int m0 = 0; m0 < 2048; m0 += 64) {
        __syncthreads();
        tile_f16_issue<4, 3>(kbase + (size_t)m0 * 64, 64, kT, ALD, t);   // 64x64
        tile_f16_issue<4, 3>(vtbase + m0, 2048, vtT, ALD, t);            // 64x64
        tile_f16_wait();
        __syncthreads();

        // S = Q K^T  (16 x 64 per wave, 4 column tiles, 2 k-slices each)
        v8f s[4];
#pragma unroll
        for (int tt = 0; tt < 4; ++tt) {
            v8f z = {};
            v16h b0 = load_b16(&kT[(tt * 16) * ALD], ALD);
            v16h b1 = load_b16(&kT[(tt * 16) * ALD + 32], ALD);
            z = WMMA_F32_F16(aq0, b0, z);
            z = WMMA_F32_F16(aq1, b1, z);
            s[tt] = z;
        }

        // Online softmax. Row (v + 8*(lane>>4)) lives in one 16-lane half:
        // xor-reduce with masks 1/2/4/8 stays within the half.
#pragma unroll
        for (int v = 0; v < 8; ++v) {
            float rmax = fmaxf(fmaxf(s[0][v], s[1][v]), fmaxf(s[2][v], s[3][v]));
            rmax = fmaxf(rmax, __shfl_xor(rmax, 1, 32));
            rmax = fmaxf(rmax, __shfl_xor(rmax, 2, 32));
            rmax = fmaxf(rmax, __shfl_xor(rmax, 4, 32));
            rmax = fmaxf(rmax, __shfl_xor(rmax, 8, 32));
            float mn   = fmaxf(mrow[v], rmax);
            float corr = __expf(mrow[v] - mn);
            mrow[v] = mn;
            float rsum = 0.f;
#pragma unroll
            for (int tt = 0; tt < 4; ++tt) {
                float p = __expf(s[tt][v] - mn);
                s[tt][v] = p;
                rsum += p;
            }
            rsum += __shfl_xor(rsum, 1, 32);
            rsum += __shfl_xor(rsum, 2, 32);
            rsum += __shfl_xor(rsum, 4, 32);
            rsum += __shfl_xor(rsum, 8, 32);
            lrow[v] = lrow[v] * corr + rsum;
#pragma unroll
            for (int tt = 0; tt < 4; ++tt) o[tt][v] *= corr;
        }

        // P: C-layout -> LDS -> A-layout for the PV WMMA.
#pragma unroll
        for (int tt = 0; tt < 4; ++tt) {
#pragma unroll
            for (int v = 0; v < 8; ++v) {
                int r = v + 8 * (lane >> 4);
                int c = tt * 16 + (lane & 15);
                pT[wave][r * ALD + c] = (_Float16)s[tt][v];
            }
        }
        __syncthreads();

        v16h ap0 = load_a16(&pT[wave][0], ALD);
        v16h ap1 = load_a16(&pT[wave][32], ALD);
#pragma unroll
        for (int tt = 0; tt < 4; ++tt) {
            v16h b0 = load_b16(&vtT[(tt * 16) * ALD], ALD);
            v16h b1 = load_b16(&vtT[(tt * 16) * ALD + 32], ALD);
            o[tt] = WMMA_F32_F16(ap0, b0, o[tt]);
            o[tt] = WMMA_F32_F16(ap1, b1, o[tt]);
        }
    }

    // Normalize and store mid[b][n][h*64 + d] as f16.
    const int b = bh >> 4, h = bh & 15;
#pragma unroll
    for (int tt = 0; tt < 4; ++tt) {
#pragma unroll
        for (int v = 0; v < 8; ++v) {
            int r = q0 + wave * 16 + v + 8 * (lane >> 4);
            int d = tt * 16 + (lane & 15);
            float val = o[tt][v] / lrow[v];
            mid[((size_t)(b * 2048 + r)) * 1024 + h * 64 + d] = (_Float16)val;
        }
    }
}

// ---------------------------------------------------------------------------
// Output projection: out[r][e] = sum_k mid[r][k] * Wo[e][k] + bo[e], fp32 out.
// Same 128x64 tiling as proj_kernel; A tile is already f16 (async-copy path).
// ---------------------------------------------------------------------------
__global__ void outproj_kernel(const _Float16* __restrict__ mid,
                               const float* __restrict__ Wo,
                               const float* __restrict__ bo,
                               float* __restrict__ out) {
    __shared__ __align__(16) _Float16 aT[128 * PLD];
    __shared__ __align__(16) _Float16 wT[64 * PLD];

    const int t    = threadIdx.x;
    const int wave = t >> 5;
    const int lane = t & 31;
    const int row0 = blockIdx.x * 128;
    const int col0 = blockIdx.y * 64;

    v8f acc[2][4] = {};

    for (int k0 = 0; k0 < 1024; k0 += 32) {
        __syncthreads();
        tile_f16_issue<4, 2>(mid + (size_t)row0 * 1024 + k0, 1024, aT, PLD, t); // 128x32
        tile_f32_to_lds<4>(Wo + (size_t)col0 * 1024 + k0, 1024, wT, t);
        tile_f16_wait();
        __syncthreads();

        v16h a0 = load_a16(&aT[(wave * 32) * PLD], PLD);
        v16h a1 = load_a16(&aT[(wave * 32 + 16) * PLD], PLD);
#pragma unroll
        for (int tt = 0; tt < 4; ++tt) {
            v16h b = load_b16(&wT[(tt * 16) * PLD], PLD);
            acc[0][tt] = WMMA_F32_F16(a0, b, acc[0][tt]);
            acc[1][tt] = WMMA_F32_F16(a1, b, acc[1][tt]);
        }
    }

    const int half = lane >> 4;
    const int nn   = lane & 15;
#pragma unroll
    for (int sub = 0; sub < 2; ++sub) {
#pragma unroll
        for (int tt = 0; tt < 4; ++tt) {
#pragma unroll
            for (int v = 0; v < 8; ++v) {
                int r = row0 + wave * 32 + sub * 16 + v + 8 * half;
                int e = col0 + tt * 16 + nn;
                out[(size_t)r * 1024 + e] = acc[sub][tt][v] + bo[e];
            }
        }
    }
}

// ---------------------------------------------------------------------------
// Launch. Inputs: x, context, Wq, Wk, Wv, Wo, bo (all fp32).
// Workspace (f16): q | k | v^T | mid, each 8,388,608 halves (16 MB), 64 MB total.
// ---------------------------------------------------------------------------
extern "C" void kernel_launch(void* const* d_in, const int* in_sizes, int n_in,
                              void* d_out, int out_size, void* d_ws, size_t ws_size,
                              hipStream_t stream) {
    (void)in_sizes; (void)n_in; (void)out_size; (void)ws_size;

    const float* x   = (const float*)d_in[0];
    const float* ctx = (const float*)d_in[1];
    const float* Wq  = (const float*)d_in[2];
    const float* Wk  = (const float*)d_in[3];
    const float* Wv  = (const float*)d_in[4];
    const float* Wo  = (const float*)d_in[5];
    const float* bo  = (const float*)d_in[6];
    float* out = (float*)d_out;

    const size_t SEG = (size_t)4 * 16 * 2048 * 64;  // 8,388,608 halves
    _Float16* ws   = (_Float16*)d_ws;
    _Float16* qb   = ws;
    _Float16* kb   = ws + SEG;
    _Float16* vtb  = ws + 2 * SEG;
    _Float16* midb = ws + 3 * SEG;

    dim3 blk(128);
    dim3 gp(64, 16);             // 8192/128 row tiles x 1024/64 col tiles

    proj_kernel<<<gp, blk, 0, stream>>>(x,   Wq, qb,  0.125f, 0);  // Q, pre-scaled
    proj_kernel<<<gp, blk, 0, stream>>>(ctx, Wk, kb,  1.0f,   0);  // K
    proj_kernel<<<gp, blk, 0, stream>>>(ctx, Wv, vtb, 1.0f,   1);  // V^T

    attn_kernel<<<dim3(32, 64), blk, 0, stream>>>(qb, kb, vtb, midb);

    outproj_kernel<<<gp, blk, 0, stream>>>(midb, Wo, bo, out);
}
